// ScaledDotProductAttention_7189775253966
// MI455X (gfx1250) — compile-verified, matrix operand order checked by reading, not run on hardware
//
#include <hip/hip_runtime.h>

// Problem constants (from reference): BS,H,S,D = 2,16,2048,64
constexpr int BS = 2;
constexpr int H  = 16;
constexpr int S  = 2048;
constexpr int D  = 64;

typedef __attribute__((ext_vector_type(16))) _Float16     v16h;
typedef __attribute__((ext_vector_type(8)))  float        v8f;
typedef __attribute__((ext_vector_type(4)))  float        f32x4;
typedef __attribute__((ext_vector_type(4)))  int          i32x4;
typedef __attribute__((ext_vector_type(8)))  int          i32x8;
typedef __attribute__((ext_vector_type(4)))  unsigned int u32x4;

#if defined(__has_builtin)
#  if __has_builtin(__builtin_amdgcn_tensor_load_to_lds) && \
      __has_builtin(__builtin_amdgcn_s_wait_tensorcnt)
#    define HAVE_TDM 1
#  endif
#endif
#ifndef HAVE_TDM
#  define HAVE_TDM 0
#endif

// ---------------------------------------------------------------------------
// Kernel 0: gate[b, h*64+d] = 1 + sigmoid( mean_s(g[b,s,:]) . Wg[i,:] + bg + bias )
// ---------------------------------------------------------------------------
__global__ __launch_bounds__(256) void k_gate(const float* __restrict__ g,
                                              const float* __restrict__ Wg,
                                              const float* __restrict__ bg,
                                              const float* __restrict__ gbias,
                                              float* __restrict__ gate)
{
    __shared__ float part[256];
    __shared__ float gavg[D];
    const int b   = blockIdx.x;
    const int tid = threadIdx.x;
    const int d   = tid & 63;
    const int sg  = tid >> 6;             // 0..3 strided partial sums over s

    const float* gb = g + (size_t)b * S * D;
    float s = 0.f;
    for (int si = sg; si < S; si += 4) s += gb[(size_t)si * D + d];
    part[tid] = s;
    __syncthreads();
    if (tid < D) {
        float t = part[tid] + part[tid + 64] + part[tid + 128] + part[tid + 192];
        gavg[tid] = t * (1.0f / (float)S);
    }
    __syncthreads();
    for (int i = tid; i < H * D; i += 256) {
        const float* w = Wg + (size_t)i * D;
        float acc = 0.f;
        #pragma unroll 8
        for (int d2 = 0; d2 < D; ++d2) acc += gavg[d2] * w[d2];
        float gv = acc + bg[i] + gbias[i];
        gate[b * (H * D) + i] = 1.0f + 1.0f / (1.0f + __expf(-gv));
    }
}

// ---------------------------------------------------------------------------
// Kernel 1: raw masked logits = (gate*Q/sqrt(D)) . (gate*K)^T  via WMMA f16.
// One wave per 16-row query strip per (b,h); 128 K tiles.
//
// ISA 16-bit A-layout (16x32): lane L (row M=L&15), hi=L>>4:
//   halves e=0..7  -> K = hi*8 + e ;  halves e=8..15 -> K = 16 + hi*8 + (e-8)
// B (32x16) symmetric with lane = N. C/D f32: acc[r] -> row (r+hi*8), col (L&15).
// Epilogue transposes the C tile through LDS so mask loads / attn stores are
// contiguous b128 per lane; attn stores are non-temporal (stream 536 MB past L2,
// keep mask/K/V resident in the 192 MB L2).
// ---------------------------------------------------------------------------
__global__ __launch_bounds__(128) void k_scores(const float* __restrict__ qg,
                                                const float* __restrict__ kg,
                                                const int*   __restrict__ mask,
                                                const float* __restrict__ gate,
                                                float* __restrict__ attn)
{
    __shared__ float stile[4][16 * 16];           // per-wave C-tile bounce buffer

    const int lane  = threadIdx.x & 31;
    const int wave  = threadIdx.x >> 5;
    const int qtile = blockIdx.x * 4 + wave;      // 0..127
    const int h     = blockIdx.y;
    const int b     = blockIdx.z;
    const int row16 = lane & 15;
    const int hi    = lane >> 4;
    const int off   = hi * 8;
    const float invT = 0.125f;                    // 1/sqrt(64)

    const size_t bh    = (size_t)(b * H + h);
    const float* qrow  = qg + (bh * S + (size_t)qtile * 16 + row16) * D;
    const float* gbase = gate + b * (H * D) + h * D;
    float*       tl    = stile[wave];

    // Per-lane gate factors for the K-dims this lane touches.
    float gk[32];
    #pragma unroll
    for (int e = 0; e < 8; ++e) {
        gk[e]      = gbase[off + e];
        gk[8 + e]  = gbase[16 + off + e];
        gk[16 + e] = gbase[32 + off + e];
        gk[24 + e] = gbase[48 + off + e];
    }

    // Gated, pre-scaled Q in A layout: two 16x32 tiles (K = 0..63).
    v16h a0, a1;
    #pragma unroll
    for (int e = 0; e < 8; ++e) {
        a0[e]     = (_Float16)(qrow[off + e]      * gk[e]      * invT);
        a0[e + 8] = (_Float16)(qrow[16 + off + e] * gk[8 + e]  * invT);
        a1[e]     = (_Float16)(qrow[32 + off + e] * gk[16 + e] * invT);
        a1[e + 8] = (_Float16)(qrow[48 + off + e] * gk[24 + e] * invT);
    }

    const float* khead = kg + bh * S * D;
    const int*   mrow  = mask + (size_t)b * S * S;    // mask[b,0,:,:]
    float*       arow  = attn + bh * S * S;
    const int    rl    = lane >> 1;                   // row-major epilogue mapping
    const int    cb    = (lane & 1) * 8;

    for (int kt = 0; kt < 128; ++kt) {
        const float* kb = khead + ((size_t)kt * 16 + row16) * D;
        if (kt + 1 < 128)   // -> global_prefetch_b8
            __builtin_prefetch(khead + ((size_t)(kt + 1) * 16 + row16) * D, 0, 1);

        v16h b0, b1;
        #pragma unroll
        for (int e = 0; e < 8; ++e) {
            b0[e]     = (_Float16)(kb[off + e]      * gk[e]);
            b0[e + 8] = (_Float16)(kb[16 + off + e] * gk[8 + e]);
            b1[e]     = (_Float16)(kb[32 + off + e] * gk[16 + e]);
            b1[e + 8] = (_Float16)(kb[48 + off + e] * gk[24 + e]);
        }

        v8f acc = {};
        acc = __builtin_amdgcn_wmma_f32_16x16x32_f16(false, a0, false, b0,
                                                     (short)0, acc, false, false);
        acc = __builtin_amdgcn_wmma_f32_16x16x32_f16(false, a1, false, b1,
                                                     (short)0, acc, false, false);

        // C-layout -> LDS -> row-major regs (intra-wave; LDS is in-order per wave).
        #pragma unroll
        for (int r = 0; r < 8; ++r)
            tl[(r + hi * 8) * 16 + row16] = acc[r];
        asm volatile("" ::: "memory");
        f32x4 s0 = *(const f32x4*)&tl[rl * 16 + cb];
        f32x4 s1 = *(const f32x4*)&tl[rl * 16 + cb + 4];
        asm volatile("" ::: "memory");

        const size_t mofs = (size_t)(qtile * 16 + rl) * S + (size_t)kt * 16 + cb;
        i32x4 m0 = *(const i32x4*)(mrow + mofs);      // mask reused 16x -> temporal
        i32x4 m1 = *(const i32x4*)(mrow + mofs + 4);
        f32x4 o0, o1;
        #pragma unroll
        for (int i = 0; i < 4; ++i) {
            o0[i] = m0[i] ? s0[i] : -1.0e9f;
            o1[i] = m1[i] ? s1[i] : -1.0e9f;
        }
        __builtin_nontemporal_store(o0, (f32x4*)(arow + mofs));
        __builtin_nontemporal_store(o1, (f32x4*)(arow + mofs + 4));
    }
}

// ---------------------------------------------------------------------------
// V-chunk staging into LDS (32 rows x 64 cols, f32).
// Preferred path: Tensor Data Mover DMA (EXEC-ignoring; one issue per wave 0).
// This toolchain declares the 6-arg form:
//   (uint32x4 g0, int32x8 g1, int32x4 g2, int32x4 g3, int32x8 g4, i32 cpol)
// ---------------------------------------------------------------------------
#if HAVE_TDM
__device__ inline void tdm_load_vtile(const float* src, unsigned int lds_byte_addr)
{
    const unsigned long long ga = (unsigned long long)(size_t)src;
    u32x4 g0 = {};
    i32x8 g1 = {};
    i32x4 g2 = {};
    i32x4 g3 = {};
    i32x8 g4 = {};
    g0[0] = 1u;                                               // count=1, user load
    g0[1] = lds_byte_addr;                                    // D#.lds_addr
    g0[2] = (unsigned int)(ga & 0xFFFFFFFFull);               // global_addr[31:0]
    g0[3] = (unsigned int)((ga >> 32) & 0x1FFFFFFull)         // global_addr[56:32]
          | (2u << 30);                                       // type = 2 (image)
    g1[0] = (int)(2u << 16);                                  // data_size = 4 bytes
    g1[1] = (int)(64u << 16);                                 // tensor_dim0 = 64
    g1[2] = (int)(32u << 16);                                 // tensor_dim1 = 32
    g1[3] = (int)(64u << 16);                                 // tile_dim0 = 64
    g1[4] = 32;                                               // tile_dim1 = 32
    g1[5] = 64;                                               // tensor_dim0_stride = 64
    __builtin_amdgcn_tensor_load_to_lds(g0, g1, g2, g3, g4, 0);   // -> TENSORcnt
}
#endif

__device__ inline void stage_vtile_fallback(const float* src, float* dstLds)
{
    // 128 threads x 16 contiguous floats = 32x64 tile, b128-coalesced.
    const int t    = threadIdx.x;
    const int row  = t >> 2;
    const int cbase = (t & 3) * 16;
    #pragma unroll
    for (int j = 0; j < 4; ++j) {
        f32x4 v = *(const f32x4*)(src + (size_t)row * D + cbase + j * 4);
        *(f32x4*)(dstLds + row * D + cbase + j * 4) = v;
    }
}

// ---------------------------------------------------------------------------
// Kernel 2: row softmax (in-place normalize of attn) fused with out = P.V
// via WMMA f16. V chunks are TDM-DMA'd into double-buffered LDS and the DMA
// is pipelined one chunk ahead (chunk 0 overlaps the whole stats phase).
// ---------------------------------------------------------------------------
__global__ __launch_bounds__(128) void k_softmax_pv(const float* __restrict__ vg,
                                                    float* __restrict__ attn,
                                                    float* __restrict__ out)
{
    __shared__ float vtile[2][32 * D];            // double-buffered V chunk

    const int lane  = threadIdx.x & 31;
    const int wave  = threadIdx.x >> 5;
    const int qtile = blockIdx.x * 4 + wave;
    const int h     = blockIdx.y;
    const int b     = blockIdx.z;
    const int row16 = lane & 15;
    const int hi    = lane >> 4;
    const int off   = hi * 8;

    const size_t bh    = (size_t)(b * H + h);
    float*       arow  = attn + bh * S * S;
    const float* vhead = vg + bh * S * D;

#if HAVE_TDM
    if (wave == 0)   // EXEC-ignoring DMA: issue once; overlaps the stats phase
        tdm_load_vtile(vhead, (unsigned int)(size_t)&vtile[0][0]);
#endif

    // Phase A: per-row max and 1/sum(exp) for the wave's 16 rows (wave32 shuffles).
    float myM = 0.f, myInv = 0.f;
    for (int r = 0; r < 16; ++r) {
        const float* rp = arow + (size_t)(qtile * 16 + r) * S;
        float mx = -3.4e38f;
        for (int j = lane; j < S; j += 32) mx = fmaxf(mx, rp[j]);
        #pragma unroll
        for (int dlt = 16; dlt >= 1; dlt >>= 1) mx = fmaxf(mx, __shfl_xor(mx, dlt, 32));
        float sum = 0.f;
        for (int j = lane; j < S; j += 32) sum += __expf(rp[j] - mx);
        #pragma unroll
        for (int dlt = 16; dlt >= 1; dlt >>= 1) sum += __shfl_xor(sum, dlt, 32);
        if (row16 == r) { myM = mx; myInv = 1.0f / sum; }
    }

    // Phase B: normalize attn in place (NT stream) and accumulate out = P.V.
    v8f acc[4] = {};
    float* prow = arow + (size_t)(qtile * 16 + row16) * S;   // this lane's A row

    for (int c = 0; c < 64; ++c) {
        const int k0 = c * 32;
#if HAVE_TDM
        if (wave == 0) __builtin_amdgcn_s_wait_tensorcnt(0);
        __syncthreads();                          // vtile[c&1] ready; buf (c+1)&1 free
        if (wave == 0 && c + 1 < 64)
            tdm_load_vtile(vhead + (size_t)(c + 1) * 32 * D,
                           (unsigned int)(size_t)&vtile[(c + 1) & 1][0]);
        const float* vt = vtile[c & 1];
#else
        __syncthreads();
        stage_vtile_fallback(vhead + (size_t)c * 32 * D, vtile[0]);
        __syncthreads();
        const float* vt = vtile[0];
#endif

        // P tile: NT read of raw logits, exp-normalize, NT write back, cvt to f16 A.
        f32x4 x0 = __builtin_nontemporal_load((const f32x4*)&prow[k0 + off]);
        f32x4 x1 = __builtin_nontemporal_load((const f32x4*)&prow[k0 + off + 4]);
        f32x4 x2 = __builtin_nontemporal_load((const f32x4*)&prow[k0 + 16 + off]);
        f32x4 x3 = __builtin_nontemporal_load((const f32x4*)&prow[k0 + 16 + off + 4]);
        f32x4 p0, p1, p2, p3;
        #pragma unroll
        for (int i = 0; i < 4; ++i) {
            p0[i] = __expf(x0[i] - myM) * myInv;
            p1[i] = __expf(x1[i] - myM) * myInv;
            p2[i] = __expf(x2[i] - myM) * myInv;
            p3[i] = __expf(x3[i] - myM) * myInv;
        }
        __builtin_nontemporal_store(p0, (f32x4*)&prow[k0 + off]);
        __builtin_nontemporal_store(p1, (f32x4*)&prow[k0 + off + 4]);
        __builtin_nontemporal_store(p2, (f32x4*)&prow[k0 + 16 + off]);
        __builtin_nontemporal_store(p3, (f32x4*)&prow[k0 + 16 + off + 4]);

        v16h pa;
        #pragma unroll
        for (int i = 0; i < 4; ++i) {
            pa[i]      = (_Float16)p0[i];
            pa[i + 4]  = (_Float16)p1[i];
            pa[i + 8]  = (_Float16)p2[i];
            pa[i + 12] = (_Float16)p3[i];
        }

        #pragma unroll
        for (int t = 0; t < 4; ++t) {
            v16h vb;
            const int col = t * 16 + row16;
            #pragma unroll
            for (int e = 0; e < 8; ++e) {
                vb[e]     = (_Float16)vt[(off + e) * D + col];
                vb[e + 8] = (_Float16)vt[(16 + off + e) * D + col];
            }
            acc[t] = __builtin_amdgcn_wmma_f32_16x16x32_f16(false, pa, false, vb,
                                                            (short)0, acc[t], false, false);
        }
    }

    float* orow = out + bh * S * D;
    #pragma unroll
    for (int t = 0; t < 4; ++t) {
        #pragma unroll
        for (int r = 0; r < 8; ++r) {
            orow[(size_t)(qtile * 16 + r + hi * 8) * D + t * 16 + row16] = acc[t][r];
        }
    }
}

// ---------------------------------------------------------------------------
// Host launch
// ---------------------------------------------------------------------------
extern "C" void kernel_launch(void* const* d_in, const int* in_sizes, int n_in,
                              void* d_out, int out_size, void* d_ws, size_t ws_size,
                              hipStream_t stream)
{
    (void)in_sizes; (void)n_in; (void)out_size; (void)ws_size;
    const float* q     = (const float*)d_in[0];
    const float* k     = (const float*)d_in[1];
    const float* v     = (const float*)d_in[2];
    const float* g     = (const float*)d_in[3];
    const int*   mask  = (const int*)d_in[4];
    const float* Wg    = (const float*)d_in[5];
    const float* bg    = (const float*)d_in[6];
    const float* gbias = (const float*)d_in[7];

    float* out  = (float*)d_out;                       // (BS,H,S,D)
    float* attn = out + (size_t)BS * H * S * D;        // (BS,H,S,S), 2nd tuple elem
    float* gate = (float*)d_ws;                        // BS*H*D floats of scratch

    k_gate<<<dim3(BS), dim3(256), 0, stream>>>(g, Wg, bg, gbias, gate);

    dim3 grid(S / 16 / 4, H, BS);                      // (32,16,2), 4 waves/block
    dim3 block(128);
    k_scores<<<grid, block, 0, stream>>>(q, k, mask, gate, attn);
    k_softmax_pv<<<grid, block, 0, stream>>>(v, attn, out);
}
